// RPN__67018669687572
// MI455X (gfx1250) — matrix operand order repaired and checked by reading, not compiled, not run on hardware
//
#include <hip/hip_runtime.h>
#include <math.h>

// ---------------------------------------------------------------------------
// Types for WMMA / TDM
// ---------------------------------------------------------------------------
typedef __attribute__((ext_vector_type(16))) _Float16 v16h;
typedef __attribute__((ext_vector_type(8)))  _Float16 v8h;
typedef __attribute__((ext_vector_type(8)))  float    v8f;
typedef __attribute__((ext_vector_type(4)))  unsigned int u32x4;
typedef __attribute__((ext_vector_type(8)))  int      i32x8;
typedef __attribute__((ext_vector_type(4)))  int      i32x4;

#define AS_STRIDE 40    // halves; 80B per A row -> 16B-aligned 8-half runs
#define BS_STRIDE 40    // halves; 80B per B row (TDM pads 16B every 64B)
#define ACT_STRIDE 264  // halves; 528B per row
#define NEGV (-1e9f)

// ---------------------------------------------------------------------------
// Kernel 1: repack conv weights -> f16, layout [tap(9)][cc(8)][n(256)][kk(32)]
// so each (tap,cc) B tile is one contiguous 16KB block (TDM-friendly) and a
// WMMA B-fragment is two contiguous b128 reads per lane.
// w_conv layout: (n=256, c=256, ky=3, kx=3) row-major.
// ---------------------------------------------------------------------------
__global__ void k_wt(const float* __restrict__ w, _Float16* __restrict__ Wt) {
  int idx = blockIdx.x * 256 + threadIdx.x;
  if (idx >= 589824) return;
  int kk  = idx & 31;
  int n   = (idx >> 5) & 255;
  int cc  = (idx >> 13) & 7;
  int tap = idx >> 16;
  int c   = cc * 32 + kk;
  Wt[idx] = (_Float16)w[(n * 256 + c) * 9 + tap];
}

__global__ void k_init(int* __restrict__ cnt) {
  if (threadIdx.x < 32) cnt[threadIdx.x] = 0;
}

// ---------------------------------------------------------------------------
// Kernel 2: fused 3x3 conv (implicit GEMM via WMMA f16) + ReLU + 1x1 heads
// + anchor decode + clip.  Block = 256 thr = 8 waves; tile 64(M) x 256(N).
// B tiles are DMA'd into LDS by the Tensor Data Mover with row padding.
// ---------------------------------------------------------------------------
__global__ __launch_bounds__(256)
void k_conv(const float* __restrict__ X, const _Float16* __restrict__ Wt,
            const float* __restrict__ bconv,
            const float* __restrict__ wobj, const float* __restrict__ bobj,
            const float* __restrict__ wdel, const float* __restrict__ bdel,
            float* __restrict__ sc_out, float* __restrict__ bx_out,
            int H, int W, int P, float stride_f,
            float aw0, float ah0, float aw1, float ah1, float aw2, float ah2)
{
  __shared__ __align__(16) _Float16 As[64 * AS_STRIDE];
  __shared__ __align__(16) _Float16 Bsh[256 * BS_STRIDE];
  __shared__ __align__(16) _Float16 Act[64 * ACT_STRIDE];
  __shared__ float O15[64 * 16];
  __shared__ int mY[64], mX[64], mB[64];

  const int t    = threadIdx.x;
  const int wave = t >> 5;
  const int lane = t & 31;
  const int mbase = blockIdx.x * 64;
  const int HW = H * W;

  if (t < 64) {
    int p = mbase + t;
    if (p < P) {
      int b = p / HW;
      int r = p - b * HW;
      int y = r / W;
      mB[t] = b; mY[t] = y; mX[t] = r - y * W;
    } else { mB[t] = -1; mY[t] = 0; mX[t] = 0; }
  }
  __syncthreads();

  // Per-thread invariant staging coordinates: mm = t&63 for every it-step.
  const int mm_t = t & 63;
  const int c0   = t >> 6;            // channel-in-chunk base: 0..3 (+4*it)
  const int myB  = mB[mm_t];
  const int myY  = mY[mm_t];
  const int myX  = mX[mm_t];

  const int wm    = wave & 3;          // 16-row M sub-tile
  const int nbase = (wave >> 2) * 128; // N half (0 or 128)
  const int lo16  = lane & 15;
  const int hi    = lane >> 4;

  // TDM descriptor constants (uniform): 1D tile, 8192 f16 elems, pad 4 DWORDs
  // every 16 DWORDs -> LDS rows at 80B stride.
  const unsigned lds_b = (unsigned)(unsigned long long)(uintptr_t)&Bsh[0];

  v8f acc[8];
  #pragma unroll
  for (int i = 0; i < 8; ++i)
    #pragma unroll
    for (int r = 0; r < 8; ++r) acc[i][r] = 0.f;

  for (int tap = 0; tap < 9; ++tap) {
    const int ky = tap / 3 - 1, kx = tap % 3 - 1;
    const int yy = myY + ky, xx = myX + kx;
    const bool okrow = (myB >= 0) & (yy >= 0) & (yy < H) & (xx >= 0) & (xx < W);
    const float* xp =
        X + (((size_t)(myB < 0 ? 0 : myB) * 256 + c0) * H + (okrow ? yy : 0)) * W
          + (okrow ? xx : 0);
    for (int cc = 0; cc < 8; ++cc) {
      // ---- TDM: DMA B tile (16KB, contiguous) into padded LDS ----
      if (t < 32) {
        unsigned long long ga =
            (unsigned long long)(uintptr_t)(Wt + (size_t)(tap * 8 + cc) * 8192);
        u32x4 g0;
        g0[0] = 1u;                                   // count=1 descriptor
        g0[1] = lds_b;                                // lds_addr (bytes)
        g0[2] = (unsigned)ga;                         // global_addr[31:0]
        g0[3] = (unsigned)(ga >> 32) | 0x80000000u;   // addr[56:32] | type=2
        i32x8 g1;
        g1[0] = 0x06D10000;   // data_size=2B, pad_en, interval=16dw, amount=4dw
        g1[1] = 0x20000000;   // tensor_dim0 = 8192 (low 16 in [31:16])
        g1[2] = 0x00010000;   // tensor_dim0 hi=0, tensor_dim1 = 1
        g1[3] = 0x20000000;   // tensor_dim1 hi=0, tile_dim0 = 8192
        g1[4] = 1;            // tile_dim1 = 1, tile_dim2 = 0
        g1[5] = 8192;         // tensor_dim0_stride low 32
        g1[6] = 0x20000000;   // stride hi=0, tensor_dim1_stride low16
        g1[7] = 0;
        i32x4 z4; z4[0] = 0; z4[1] = 0; z4[2] = 0; z4[3] = 0;
        i32x8 z8;
        #pragma unroll
        for (int i = 0; i < 8; ++i) z8[i] = 0;
        __builtin_amdgcn_tensor_load_to_lds(g0, g1, z4, z4, z8, 0);
      }

      // ---- stage A tile [64 m][32 c] into LDS (coalesced along m) ----
      const float* xc = xp + (size_t)(cc * 32) * HW;
      #pragma unroll
      for (int it = 0; it < 8; ++it) {
        float v = okrow ? xc[(size_t)(4 * it) * HW] : 0.f;
        As[mm_t * AS_STRIDE + c0 + 4 * it] = (_Float16)v;
      }

      if (t < 32) __builtin_amdgcn_s_wait_tensorcnt((unsigned short)0);
      __syncthreads();

      // ---- A fragment (16-bit A 16x32 layout: lane holds k runs of 8) ----
      const int mrow = wm * 16 + lo16;
      v8h a0 = *(const v8h*)&As[mrow * AS_STRIDE + hi * 8];
      v8h a1 = *(const v8h*)&As[mrow * AS_STRIDE + 16 + hi * 8];
      v16h afrag;
      #pragma unroll
      for (int i = 0; i < 8; ++i) { afrag[i] = a0[i]; afrag[8 + i] = a1[i]; }

      // ---- preload all 8 B fragments (clause of ds_load_b128), then issue
      //      the 8 WMMAs back-to-back so the matrix pipe isn't serialized
      //      behind per-tile DScnt drains ----
      v16h bfrag[8];
      #pragma unroll
      for (int tn = 0; tn < 8; ++tn) {
        int n = nbase + tn * 16 + lo16;
        const _Float16* bp = &Bsh[n * BS_STRIDE + hi * 8];
        v8h b0 = *(const v8h*)bp;
        v8h b1 = *(const v8h*)(bp + 16);
        #pragma unroll
        for (int i = 0; i < 8; ++i) { bfrag[tn][i] = b0[i]; bfrag[tn][8 + i] = b1[i]; }
      }
      #pragma unroll
      for (int tn = 0; tn < 8; ++tn) {
        acc[tn] = __builtin_amdgcn_wmma_f32_16x16x32_f16(
            false, afrag, false, bfrag[tn], (short)0, acc[tn], false, false);
      }
      __syncthreads();
    }
  }

  // ---- epilogue: bias + ReLU -> Act tile (f16) in LDS ----
  #pragma unroll
  for (int tn = 0; tn < 8; ++tn) {
    int n = nbase + tn * 16 + lo16;
    float bias = bconv[n];
    #pragma unroll
    for (int r = 0; r < 8; ++r) {
      int m = wm * 16 + hi * 8 + r;   // C/D layout: VGPR r -> M=r (+8 hi lanes)
      float v = acc[tn][r] + bias;
      Act[m * ACT_STRIDE + n] = (_Float16)(v > 0.f ? v : 0.f);
    }
  }
  __syncthreads();

  // ---- 1x1 heads: 15 outputs (3 logits + 12 deltas) per position ----
  for (int task = t; task < 64 * 15; task += 256) {
    int m = task / 15, j = task - (task / 15) * 15;
    const float* w1 = (j < 3) ? (wobj + j * 256) : (wdel + (j - 3) * 256);
    float s = (j < 3) ? bobj[j] : bdel[j - 3];
    for (int n = 0; n < 256; ++n) s += (float)Act[m * ACT_STRIDE + n] * w1[n];
    O15[m * 16 + j] = s;
  }
  __syncthreads();

  // ---- anchor decode + clip + validity ----
  const float AW[3] = {aw0, aw1, aw2};
  const float AH[3] = {ah0, ah1, ah2};
  const float LOG_MAX = 4.1351666f;
  for (int task = t; task < 192; task += 256) {
    int m = task / 3, a = task - (task / 3) * 3;
    int b = mB[m];
    if (b < 0) continue;
    int y = mY[m], x = mX[m];
    float wa = AW[a], ha = AH[a];
    float cx = x * stride_f, cy = y * stride_f;
    float dx = O15[m * 16 + 3 + a * 4 + 0];
    float dy = O15[m * 16 + 3 + a * 4 + 1];
    float dw = fminf(O15[m * 16 + 3 + a * 4 + 2], LOG_MAX);
    float dh = fminf(O15[m * 16 + 3 + a * 4 + 3], LOG_MAX);
    float pcx = dx * wa + cx, pcy = dy * ha + cy;
    float pw = expf(dw) * wa, ph = expf(dh) * ha;
    float x1 = pcx - 0.5f * pw, y1 = pcy - 0.5f * ph;
    float x2 = pcx + 0.5f * pw, y2 = pcy + 0.5f * ph;
    x1 = fminf(fmaxf(x1, 0.f), 1408.f); x2 = fminf(fmaxf(x2, 0.f), 1408.f);
    y1 = fminf(fmaxf(y1, 0.f), 704.f);  y2 = fminf(fmaxf(y2, 0.f), 704.f);
    float sc = O15[m * 16 + a];
    if (!((x2 - x1 > 0.f) && (y2 - y1 > 0.f))) sc = NEGV;
    size_t o = (size_t)b * (size_t)(HW * 3) + (size_t)((y * W + x) * 3 + a);
    sc_out[o] = sc;
    float* bo = bx_out + 4 * o;
    bo[0] = x1; bo[1] = y1; bo[2] = x2; bo[3] = y2;
  }
}

// ---------------------------------------------------------------------------
// Kernel 3: exact top-2000 threshold per (level, batch): 3-pass radix
// histogram on sortable uints (12 + 12 + 8 bits).
// ---------------------------------------------------------------------------
struct SelInfo { int off[4]; int n[4]; };

__device__ __forceinline__ unsigned int fkey(float f) {
  unsigned int u = __float_as_uint(f);
  return (u & 0x80000000u) ? ~u : (u | 0x80000000u);
}

__global__ __launch_bounds__(1024)
void k_select(const float* __restrict__ sc, SelInfo info,
              unsigned int* __restrict__ uthr, int* __restrict__ gcnt)
{
  __shared__ int hist[4096];
  __shared__ unsigned int s_pref;
  __shared__ int s_k, s_g;
  const int lvl = blockIdx.x >> 1;
  const int b   = blockIdx.x & 1;
  const int N   = info.n[lvl];
  const float* base = sc + info.off[lvl] + (size_t)b * N;

  unsigned int pref = 0;
  int K = 2000, G = 0;
  for (int pass = 0; pass < 3; ++pass) {
    for (int i = threadIdx.x; i < 4096; i += 1024) hist[i] = 0;
    __syncthreads();
    for (int i = threadIdx.x; i < N; i += 1024) {
      unsigned int u = fkey(base[i]);
      if (pass == 0)       atomicAdd(&hist[u >> 20], 1);
      else if (pass == 1) { if ((u >> 20) == pref) atomicAdd(&hist[(u >> 8) & 0xFFF], 1); }
      else                { if ((u >> 8)  == pref) atomicAdd(&hist[u & 0xFF], 1); }
    }
    __syncthreads();
    if (threadIdx.x == 0) {
      int nb = (pass == 2) ? 256 : 4096;
      int cum = 0, tbin = 0;
      for (int bin = nb - 1; bin >= 0; --bin) {
        int c = hist[bin];
        if (cum + c >= K) { tbin = bin; break; }
        cum += c;
      }
      G += cum; K -= cum;
      pref = (pref << ((pass == 2) ? 8 : 12)) | (unsigned)tbin;
      s_pref = pref; s_k = K; s_g = G;
    }
    __syncthreads();
    pref = s_pref; K = s_k; G = s_g;
  }
  if (threadIdx.x == 0) { uthr[blockIdx.x] = pref; gcnt[blockIdx.x] = G; }
}

// ---------------------------------------------------------------------------
// Kernel 4: compaction — gather the exact top-2000 set per (level,b) into
// concatenated [b][8000] arrays (order within a level is irrelevant; NMS
// re-sorts the full 8000).
// ---------------------------------------------------------------------------
__global__ void k_compact(const float* __restrict__ sc, const float* __restrict__ bx,
                          int N, int lvl,
                          const unsigned int* __restrict__ uthr,
                          const int* __restrict__ gcnt, int* __restrict__ cnt,
                          float* __restrict__ sc_cat, float* __restrict__ bx_cat)
{
  int gid = blockIdx.x * blockDim.x + threadIdx.x;
  if (gid >= 2 * N) return;
  int b = gid / N, i = gid - b * N;
  int sel = lvl * 2 + b;
  float f = sc[(size_t)b * N + i];
  unsigned int u = fkey(f);
  unsigned int thr = uthr[sel];
  int G = gcnt[sel];
  int pos = -1;
  if (u > thr) pos = atomicAdd(&cnt[sel * 2], 1);
  else if (u == thr) {
    int e = atomicAdd(&cnt[sel * 2 + 1], 1);
    if (e < 2000 - G) pos = G + e;
  }
  if (pos >= 0) {
    int dst = b * 8000 + lvl * 2000 + pos;
    sc_cat[dst] = f;
    const float* s4 = bx + 4 * ((size_t)b * N + i);
    float* d4 = bx_cat + 4 * (size_t)dst;
    d4[0] = s4[0]; d4[1] = s4[1]; d4[2] = s4[2]; d4[3] = s4[3];
  }
}

// ---------------------------------------------------------------------------
// Kernel 5: per-batch bitonic sort of 8000 scores (descending) in LDS.
// ---------------------------------------------------------------------------
__global__ __launch_bounds__(1024)
void k_sort(const float* __restrict__ sc_cat, int* __restrict__ sidx,
            float* __restrict__ ssc)
{
  __shared__ float key[8192];
  __shared__ unsigned short idx[8192];
  const int b = blockIdx.x;
  for (int i = threadIdx.x; i < 8192; i += 1024) {
    key[i] = (i < 8000) ? sc_cat[b * 8000 + i] : -3.0e38f;
    idx[i] = (unsigned short)i;
  }
  __syncthreads();
  for (int k = 2; k <= 8192; k <<= 1) {
    for (int j = k >> 1; j > 0; j >>= 1) {
      for (int i = threadIdx.x; i < 8192; i += 1024) {
        int ixj = i ^ j;
        if (ixj > i) {
          bool up = ((i & k) == 0);               // "up" blocks sort descending
          float a = key[i], c = key[ixj];
          bool sw = up ? (a < c) : (a > c);
          if (sw) {
            key[i] = c; key[ixj] = a;
            unsigned short ti = idx[i]; idx[i] = idx[ixj]; idx[ixj] = ti;
          }
        }
      }
      __syncthreads();
    }
  }
  for (int i = threadIdx.x; i < 8000; i += 1024) {
    sidx[b * 8000 + i] = idx[i];
    ssc[b * 8000 + i]  = key[i];
  }
}

// ---------------------------------------------------------------------------
// Kernel 6: 64x64 IoU suppression bitmask (upper triangle), boxes shifted by
// level*4096 so cross-level boxes never overlap (batched-NMS trick).
// ---------------------------------------------------------------------------
__global__ __launch_bounds__(64)
void k_iou(const float* __restrict__ bx_cat, const int* __restrict__ sidx,
           unsigned long long* __restrict__ mask)
{
  const int rb = blockIdx.x, cb = blockIdx.y, b = blockIdx.z;
  if (cb < rb) return;
  __shared__ float cbox[64][4];
  const int tid = threadIdx.x;
  {
    int cj = cb * 64 + tid;
    int src = sidx[b * 8000 + cj];
    float off = (float)(src / 2000) * 4096.0f;
    const float* p = bx_cat + 4 * ((size_t)b * 8000 + src);
    cbox[tid][0] = p[0] + off; cbox[tid][1] = p[1] + off;
    cbox[tid][2] = p[2] + off; cbox[tid][3] = p[3] + off;
  }
  __syncthreads();
  const int ri = rb * 64 + tid;
  int srcr = sidx[b * 8000 + ri];
  float offr = (float)(srcr / 2000) * 4096.0f;
  const float* pr = bx_cat + 4 * ((size_t)b * 8000 + srcr);
  float rx1 = pr[0] + offr, ry1 = pr[1] + offr;
  float rx2 = pr[2] + offr, ry2 = pr[3] + offr;
  float ra = (rx2 - rx1) * (ry2 - ry1);
  unsigned long long bits = 0ull;
  for (int j = 0; j < 64; ++j) {
    if (cb * 64 + j <= ri) continue;               // only suppress later boxes
    float ix1 = fmaxf(rx1, cbox[j][0]);
    float iy1 = fmaxf(ry1, cbox[j][1]);
    float ix2 = fminf(rx2, cbox[j][2]);
    float iy2 = fminf(ry2, cbox[j][3]);
    float iw = fmaxf(ix2 - ix1, 0.f), ih = fmaxf(iy2 - iy1, 0.f);
    float inter = iw * ih;
    float ca = (cbox[j][2] - cbox[j][0]) * (cbox[j][3] - cbox[j][1]);
    float uni = ra + ca - inter;
    float iou = (uni > 0.f) ? inter / uni : 0.f;
    if (iou > 0.7f) bits |= (1ull << j);
  }
  mask[((size_t)b * 8000 + ri) * 125 + cb] = bits;
}

// ---------------------------------------------------------------------------
// Kernel 7: greedy NMS reduce (chunked sequential) + final output write.
// ---------------------------------------------------------------------------
__global__ __launch_bounds__(128)
void k_reduce(const float* __restrict__ bx_cat, const int* __restrict__ sidx,
              const float* __restrict__ ssc,
              const unsigned long long* __restrict__ mask,
              float* __restrict__ out)
{
  __shared__ unsigned long long remv[125];
  __shared__ int keeplist[1000];
  __shared__ int s_nk;
  __shared__ unsigned long long s_kb;
  const int b = blockIdx.x, t = threadIdx.x;
  for (int i = t; i < 125; i += 128) remv[i] = 0ull;
  if (t == 0) s_nk = 0;
  __syncthreads();

  for (int c = 0; c < 125; ++c) {
    if (t == 0) {
      unsigned long long w = remv[c];
      unsigned long long kb = 0ull;
      int base = c * 64;
      for (int r = 0; r < 64 && s_nk < 1000; ++r) {
        int i = base + r;
        if ((w >> r) & 1ull) continue;
        if (ssc[b * 8000 + i] <= -1e8f) continue;   // invalid
        kb |= (1ull << r);
        keeplist[s_nk++] = i;
        w |= mask[((size_t)b * 8000 + i) * 125 + c];
      }
      remv[c] = w;
      s_kb = kb;
    }
    __syncthreads();
    unsigned long long kb = s_kb;
    while (kb) {
      int r = __builtin_ctzll(kb);
      kb &= kb - 1ull;
      int i = c * 64 + r;
      for (int w = c + 1 + t; w < 125; w += 128)
        remv[w] |= mask[((size_t)b * 8000 + i) * 125 + w];
    }
    __syncthreads();
    if (s_nk >= 1000) break;
  }
  __syncthreads();
  const int nkeep = s_nk;
  for (int k = t; k < 1000; k += 128) {
    float* o = out + ((size_t)b * 1000 + k) * 5;
    if (k < nkeep) {
      int i = keeplist[k];
      int src = sidx[b * 8000 + i];
      const float* p = bx_cat + 4 * ((size_t)b * 8000 + src);
      o[0] = p[0]; o[1] = p[1]; o[2] = p[2]; o[3] = p[3];
      o[4] = ssc[b * 8000 + i];
    } else {
      o[0] = 0.f; o[1] = 0.f; o[2] = 0.f; o[3] = 0.f; o[4] = NEGV;
    }
  }
}

// ---------------------------------------------------------------------------
// Host launcher
// ---------------------------------------------------------------------------
extern "C" void kernel_launch(void* const* d_in, const int* in_sizes, int n_in,
                              void* d_out, int out_size, void* d_ws, size_t ws_size,
                              hipStream_t stream) {
  (void)in_sizes; (void)n_in; (void)out_size; (void)ws_size;
  const float* feat[4] = {(const float*)d_in[0], (const float*)d_in[1],
                          (const float*)d_in[2], (const float*)d_in[3]};
  const float* w_conv = (const float*)d_in[4];
  const float* b_conv = (const float*)d_in[5];
  const float* w_obj  = (const float*)d_in[6];
  const float* b_obj  = (const float*)d_in[7];
  const float* w_del  = (const float*)d_in[8];
  const float* b_del  = (const float*)d_in[9];

  const int HH[4] = {176, 88, 44, 22};
  const int WW[4] = {352, 176, 88, 44};
  const float STR[4] = {4.f, 8.f, 16.f, 32.f};
  const float SIZ[4] = {32.f, 64.f, 128.f, 256.f};
  const float RAT[3] = {0.5f, 1.f, 2.f};
  const int B = 2;

  int Nl[4], offf[4];
  int cum = 0;
  for (int l = 0; l < 4; ++l) {
    Nl[l] = HH[l] * WW[l] * 3;
    offf[l] = cum;
    cum += B * Nl[l];
  }

  // workspace layout
  char* ws = (char*)d_ws;
  size_t o = 0;
  auto alloc = [&](size_t bytes) -> size_t {
    o = (o + 255) & ~(size_t)255;
    size_t r = o; o += bytes; return r;
  };
  _Float16* Wt       = (_Float16*)(ws + alloc((size_t)589824 * 2));
  float* sc_lvl      = (float*)   (ws + alloc((size_t)cum * 4));
  float* bx_lvl      = (float*)   (ws + alloc((size_t)cum * 16));
  unsigned int* uthr = (unsigned int*)(ws + alloc(8 * 4));
  int* gcnt          = (int*)     (ws + alloc(8 * 4));
  int* cnt           = (int*)     (ws + alloc(32 * 4));
  float* sc_cat      = (float*)   (ws + alloc((size_t)B * 8000 * 4));
  float* bx_cat      = (float*)   (ws + alloc((size_t)B * 8000 * 16));
  int* sidx          = (int*)     (ws + alloc((size_t)B * 8000 * 4));
  float* ssc         = (float*)   (ws + alloc((size_t)B * 8000 * 4));
  unsigned long long* mask =
      (unsigned long long*)(ws + alloc((size_t)B * 8000 * 125 * 8));

  // 1) weight repack + counter init
  k_wt<<<2304, 256, 0, stream>>>(w_conv, Wt);
  k_init<<<1, 32, 0, stream>>>(cnt);

  // 2) fused conv + heads + decode, per level
  for (int l = 0; l < 4; ++l) {
    int H = HH[l], W = WW[l];
    int P = B * H * W;
    float aw[3], ah[3];
    for (int a = 0; a < 3; ++a) {
      aw[a] = SIZ[l] * sqrtf(1.f / RAT[a]);
      ah[a] = SIZ[l] * sqrtf(RAT[a]);
    }
    k_conv<<<(P + 63) / 64, 256, 0, stream>>>(
        feat[l], Wt, b_conv, w_obj, b_obj, w_del, b_del,
        sc_lvl + offf[l], bx_lvl + (size_t)4 * offf[l],
        H, W, P, STR[l], aw[0], ah[0], aw[1], ah[1], aw[2], ah[2]);
  }

  // 3) exact top-2000 thresholds per (level, batch)
  SelInfo si;
  for (int l = 0; l < 4; ++l) { si.off[l] = offf[l]; si.n[l] = Nl[l]; }
  k_select<<<8, 1024, 0, stream>>>(sc_lvl, si, uthr, gcnt);

  // 4) compaction into concatenated [b][8000] arrays
  for (int l = 0; l < 4; ++l) {
    int total = 2 * Nl[l];
    k_compact<<<(total + 255) / 256, 256, 0, stream>>>(
        sc_lvl + offf[l], bx_lvl + (size_t)4 * offf[l], Nl[l], l,
        uthr, gcnt, cnt, sc_cat, bx_cat);
  }

  // 5) sort, 6) IoU bitmask, 7) greedy reduce + output
  k_sort<<<B, 1024, 0, stream>>>(sc_cat, sidx, ssc);
  k_iou<<<dim3(125, 125, B), 64, 0, stream>>>(bx_cat, sidx, mask);
  k_reduce<<<B, 128, 0, stream>>>(bx_cat, sidx, ssc, mask, (float*)d_out);
}
